// TemporalProj_20779051778732
// MI455X (gfx1250) — compile-verified, matrix operand order checked by reading, not run on hardware
//
#include <hip/hip_runtime.h>
#include <hip/hip_bf16.h>

// ---------------------------------------------------------------------------
// MoE temporal projection for MI455X (gfx1250, wave32, WMMA).
//  x  : [B=64, D=512, V=512] f32   (tokens are (b,v), feature dim D)
//  Wg : [E=8, D=512] f32           gating
//  We : [E=8, O=720, D=512] f32    experts (row-major: row = output col)
//  be : [E=8, O=720] f32
//  out: [B=64, O=720, V=512] f32
//
// Pipeline:
//  k0: We f32 -> bf16 once into d_ws (L2-resident, removes per-tile cvt)
//  k1: gating softmax + top-2 -> masked coef[token][8] in d_ws
//  k2: per 16-token tile: A (16x512 bf16) staged once in LDS, 4 waves sweep
//      all 45 N-tiles; per K-step one A fragment feeds 8 expert WMMAs.
// ---------------------------------------------------------------------------

typedef __bf16 bf16_t;
typedef __attribute__((ext_vector_type(16))) __bf16 v16bf;
typedef __attribute__((ext_vector_type(8)))  __bf16 v8bf;
typedef __attribute__((ext_vector_type(4)))  __bf16 v4bf;
typedef __attribute__((ext_vector_type(8)))  float  v8f;

#define B_BATCH 64
#define D_IN    512
#define V_VARS  512
#define O_OUT   720
#define E_EXP   8
#define NTILES  45           // ceil(720 / 16)
#define WAVES_PER_BLK 4

// ---------------------------------------------------------------------------
// Kernel 0: one-time We f32 -> bf16 conversion (row-major [E][O][D]).
// ---------------------------------------------------------------------------
__global__ __launch_bounds__(256)
void convert_we_kernel(const float* __restrict__ We, bf16_t* __restrict__ Webf)
{
    const int i = (blockIdx.x * 256 + threadIdx.x) * 4;   // n divisible by 4
    const float4 f = *(const float4*)(We + i);
    v4bf o;
    o[0] = (bf16_t)f.x; o[1] = (bf16_t)f.y;
    o[2] = (bf16_t)f.z; o[3] = (bf16_t)f.w;
    *(v4bf*)(Webf + i) = o;
}

// ---------------------------------------------------------------------------
// Kernel 1: gating.  16 tokens per block (all share the same b since V%16==0).
// Produces coef[token][e] = softmax weight if e in top-2 else 0.
// ---------------------------------------------------------------------------
__global__ __launch_bounds__(256)
void moe_gate_topk_kernel(const float* __restrict__ x,
                          const float* __restrict__ Wg,
                          float* __restrict__ coef)
{
    __shared__ float wgLds[E_EXP][D_IN];          // 16 KB
    __shared__ float part[16][16][E_EXP];         // 8 KB  [dgroup][vloc][e]
    __shared__ float logit[16][E_EXP];

    const int tid = threadIdx.x;

    for (int i = tid; i < E_EXP * D_IN; i += 256)
        wgLds[i / D_IN][i % D_IN] = Wg[i];
    __syncthreads();

    const int t0 = blockIdx.x * 16;
    const int b  = t0 / V_VARS;
    const int v0 = t0 % V_VARS;
    const float* xb = x + (size_t)b * D_IN * V_VARS + v0;

    const int vloc = tid & 15;      // token within tile (coalesced over V)
    const int dg   = tid >> 4;      // 16 d-groups

    float acc[E_EXP];
#pragma unroll
    for (int e = 0; e < E_EXP; ++e) acc[e] = 0.0f;

    for (int i = 0; i < D_IN / 16; ++i) {
        const int d = dg + 16 * i;
        const float xv = xb[(size_t)d * V_VARS + vloc];
#pragma unroll
        for (int e = 0; e < E_EXP; ++e) acc[e] += xv * wgLds[e][d];
    }
#pragma unroll
    for (int e = 0; e < E_EXP; ++e) part[dg][vloc][e] = acc[e];
    __syncthreads();

    if (tid < 16 * E_EXP) {                 // 128 threads: (e, vloc)
        const int e = tid >> 4, vv = tid & 15;
        float s = 0.0f;
#pragma unroll
        for (int g = 0; g < 16; ++g) s += part[g][vv][e];
        logit[vv][e] = s;
    }
    __syncthreads();

    if (tid < 16) {                          // one thread per token
        float l[E_EXP];
        float mx = -3.0e38f;
#pragma unroll
        for (int e = 0; e < E_EXP; ++e) { l[e] = logit[tid][e]; mx = fmaxf(mx, l[e]); }
        float sum = 0.0f;
#pragma unroll
        for (int e = 0; e < E_EXP; ++e) { l[e] = __expf(l[e] - mx); sum += l[e]; }
        const float inv = 1.0f / sum;
#pragma unroll
        for (int e = 0; e < E_EXP; ++e) l[e] *= inv;

        int   i1 = 0;  float w1 = l[0];
#pragma unroll
        for (int e = 1; e < E_EXP; ++e) if (l[e] > w1) { w1 = l[e]; i1 = e; }
        int   i2 = -1; float w2 = -1.0f;
#pragma unroll
        for (int e = 0; e < E_EXP; ++e) if (e != i1 && l[e] > w2) { w2 = l[e]; i2 = e; }

        float* cp = coef + (size_t)(t0 + tid) * E_EXP;
#pragma unroll
        for (int e = 0; e < E_EXP; ++e)
            cp[e] = (e == i1) ? w1 : ((e == i2) ? w2 : 0.0f);
    }
}

// ---------------------------------------------------------------------------
// Kernel 2: expert GEMMs + combine.  Block = 16 tokens, 4 waves sweep all
// 45 N-tiles (A tile in LDS loaded ONCE per 16 tokens).  Per K-step one A
// fragment feeds 8 expert WMMAs (8 f32 accumulators).
// ---------------------------------------------------------------------------
__global__ __launch_bounds__(128)
void moe_expert_wmma_kernel(const float* __restrict__ x,
                            const bf16_t* __restrict__ Webf,
                            const float* __restrict__ be,
                            const float* __restrict__ coef,
                            float* __restrict__ out)
{
    __shared__ __align__(16) bf16_t As[16][D_IN + 8];     // 16.6 KB, padded rows
    __shared__ float coefLds[16][E_EXP];                  // 512 B
    __shared__ float stage[WAVES_PER_BLK][16][17];        // 4.4 KB, store bounce

    const int tid = threadIdx.x;
    const int t0  = blockIdx.x * 16;
    const int b   = t0 / V_VARS;
    const int v0  = t0 % V_VARS;
    const float* xb = x + (size_t)b * D_IN * V_VARS + v0;

    // Cooperative A load: element (k, m) -> As[m][k], coalesced over m (=v).
    for (int idx = tid; idx < 16 * D_IN; idx += 128) {
        const int m = idx & 15;
        const int k = idx >> 4;
        As[m][k] = (bf16_t)xb[(size_t)k * V_VARS + m];
    }
    // Gate coefficients for the 16 tokens.
    if (tid < 16 * E_EXP)
        coefLds[tid >> 3][tid & 7] =
            coef[(size_t)(t0 + (tid >> 3)) * E_EXP + (tid & 7)];
    __syncthreads();

    const int wave = tid >> 5;
    const int lane = tid & 31;
    const int half = lane >> 4;                  // K-chunk selector (ISA layout)
    const int nn   = lane & 15;                  // B/C column owned by lane
    const int mrow = lane & 15;                  // A row owned by lane

    const size_t OD = (size_t)O_OUT * D_IN;
    float* ob_base = out + (size_t)b * O_OUT * V_VARS + v0;

    for (int nTile = wave; nTile < NTILES; nTile += WAVES_PER_BLK) {
        const int n0 = nTile * 16;

        v8f c[E_EXP];
#pragma unroll
        for (int e = 0; e < E_EXP; ++e) c[e] = (v8f)(0.0f);

        const bf16_t* wrow0 = Webf + (size_t)(n0 + nn) * D_IN;  // expert 0 row

        for (int kb = 0; kb < D_IN; kb += 32) {
            // A fragment (shared across all 8 experts): two 16B LDS loads.
            const v8bf alo = *(const v8bf*)&As[mrow][kb + half * 8];
            const v8bf ahi = *(const v8bf*)&As[mrow][kb + 16 + half * 8];
            v16bf a;
#pragma unroll
            for (int j = 0; j < 8; ++j) { a[j] = alo[j]; a[8 + j] = ahi[j]; }

#pragma unroll
            for (int e = 0; e < E_EXP; ++e) {
                // B fragment: two 16B bf16 global loads (L2-resident Webf).
                const bf16_t* wr = wrow0 + e * OD + kb + half * 8;
                const v8bf b0 = *(const v8bf*)(wr);
                const v8bf b1 = *(const v8bf*)(wr + 16);
                v16bf bm;
#pragma unroll
                for (int j = 0; j < 8; ++j) { bm[j] = b0[j]; bm[8 + j] = b1[j]; }

                c[e] = __builtin_amdgcn_wmma_f32_16x16x32_bf16(
                           /*neg_a=*/false, a, /*neg_b=*/false, bm,
                           /*c_mod=*/(short)0, c[e],
                           /*reuse_a=*/false, /*reuse_b=*/false);
            }
        }

        // combine: out[m][n] = sum_e coef[m][e] * (gemm_e + be[e][n])
        float bvals[E_EXP];
#pragma unroll
        for (int e = 0; e < E_EXP; ++e) bvals[e] = be[(size_t)e * O_OUT + n0 + nn];

#pragma unroll
        for (int r = 0; r < 8; ++r) {
            const int mr = r + 8 * half;            // C row for this VGPR slot
            float s = 0.0f;
#pragma unroll
            for (int e = 0; e < E_EXP; ++e)
                s += coefLds[mr][e] * (c[e][r] + bvals[e]);
            stage[wave][mr][nn] = s;                // tile transpose via LDS
        }

        asm volatile("s_wait_dscnt 0" ::: "memory");  // same-wave LDS RAW

        // Coalesced store: 16 consecutive lanes sweep 16 consecutive v's.
        float* ob = ob_base + (size_t)n0 * V_VARS;
#pragma unroll
        for (int it = 0; it < 8; ++it) {
            const int orow = half + 2 * it;         // output col within tile
            ob[(size_t)orow * V_VARS + nn] = stage[wave][nn][orow];
        }
    }
}

// ---------------------------------------------------------------------------
extern "C" void kernel_launch(void* const* d_in, const int* in_sizes, int n_in,
                              void* d_out, int out_size, void* d_ws, size_t ws_size,
                              hipStream_t stream)
{
    const float* x  = (const float*)d_in[0];   // [64,512,512]
    const float* Wg = (const float*)d_in[1];   // [8,512]
    const float* We = (const float*)d_in[2];   // [8,720,512]
    const float* be = (const float*)d_in[3];   // [8,720]
    // d_in[4] = top_k (==2), baked into the top-2 selection above.

    // Workspace layout: [0, 1MB) coef f32 [32768][8]; [1MB, ~7MB) Webf bf16.
    float*  coef = (float*)d_ws;
    bf16_t* Webf = (bf16_t*)((char*)d_ws + (size_t)(32768 * 8 * 4));

    float* out = (float*)d_out;                // [64,720,512]

    const int tokens     = B_BATCH * V_VARS;   // 32768
    const int tokenTiles = tokens / 16;        // 2048
    const int weElems    = E_EXP * O_OUT * D_IN;  // 2,949,120 (divisible by 1024)

    convert_we_kernel<<<dim3(weElems / 1024), dim3(256), 0, stream>>>(We, Webf);
    moe_gate_topk_kernel<<<dim3(tokenTiles), dim3(256), 0, stream>>>(x, Wg, coef);
    moe_expert_wmma_kernel<<<dim3(tokenTiles), dim3(128), 0, stream>>>(
        x, Webf, be, coef, out);
}
